// ViTEncoder_60722247631284
// MI455X (gfx1250) — compile-verified, matrix operand order checked by reading, not run on hardware
//
#include <hip/hip_runtime.h>
#include <hip/hip_bf16.h>
#include <stdint.h>

// ---------------------------------------------------------------------------
// ViT encoder forward for MI455X (gfx1250):
//   bf16 WMMA GEMMs + TDM (tensor_load_to_lds) double-buffered tile pipeline
// ---------------------------------------------------------------------------

typedef unsigned short u16;
typedef __attribute__((ext_vector_type(16))) __bf16 bf16x16;
typedef __attribute__((ext_vector_type(8)))  float  f32x8;
typedef __attribute__((ext_vector_type(4)))  unsigned int u32x4;
typedef __attribute__((ext_vector_type(8)))  int  i32x8;
typedef __attribute__((ext_vector_type(4)))  int  i32x4;

#define D_MODEL 1024
#define N_SEQ   65
#define B_BATCH 256
#define ROWS    (B_BATCH * N_SEQ)     // 16640 = 130 * 128
#define HEADS   16
#define HD      64
#define FF      4096
#define DEPTH   12
#define NPATCH  64
#define PROWS   (B_BATCH * NPATCH)    // 16384

__device__ __forceinline__ u16 f2bf(float f) {
  unsigned int x = __float_as_uint(f);
  x += 0x7fffu + ((x >> 16) & 1u);     // round-to-nearest-even
  return (u16)(x >> 16);
}

__device__ __forceinline__ bf16x16 load_frag16(const u16* p) {
  union { bf16x16 v; uint4 q[2]; } u;
  u.q[0] = *(const uint4*)(p);         // K block 0 (8 bf16, 16B)
  u.q[1] = *(const uint4*)(p + 16);    // K block +16 (8 bf16, 16B)
  return u.v;
}

// ------------------------------- GEMM --------------------------------------
// C[M x Nn] = A[M x K] (bf16, row major) * Bt[Nn x K] (bf16, pre-transposed)
// Tiles are pulled into LDS by the Tensor Data Mover (double buffered);
// epilogue: +bias, optional exact GELU, optional residual add (fp32 out),
// optional bf16 out, optional patch-row remap (row -> b*65 + p + 1).
#define BM 128
#define BN 256
#define BKK 32
#define LSTR 40   // 32 + 8 pad bf16 (80B rows; TDM pad 4 dwords / 16 dwords)

// Build a 2D D# (ISA 8.3/8.4) and issue TENSOR_LOAD_TO_LDS:
//   tile = rows x 32 bf16 elements, row stride K elements, LDS rows padded
//   by 16B every 64B so LDS row stride = 80B = LSTR*2.
__device__ __forceinline__ void tdm_load_tile(const u16* gbase, unsigned ldsOff,
                                              int rows, int K,
                                              long rowStart, int k0)
{
  unsigned long long ga = (unsigned long long)(uintptr_t)gbase +
      ((unsigned long long)rowStart * (unsigned long long)K +
       (unsigned long long)k0) * 2ull;
  u32x4 g0;
  g0[0] = 1u;                                       // count=1 (valid user D#)
  g0[1] = ldsOff;                                   // lds_addr (bytes)
  g0[2] = (unsigned)(ga & 0xffffffffu);             // global_addr[31:0]
  g0[3] = (unsigned)((ga >> 32) & 0x01ffffffu)      // global_addr[56:32]
        | (2u << 30);                               // type = 2 ("image")
  i32x8 g1;
  g1[0] = (int)((1u << 16)      // data_size = 2 bytes (bf16)
              | (1u << 20)      // pad_enable
              | (3u << 22)      // pad_interval: 16 DWORDs (64B = one tile row)
              | (3u << 25));    // pad_amount: 4 DWORDs (16B) -> 80B rows
  g1[1] = (int)(((unsigned)K & 0xffffu) << 16);     // tensor_dim0[15:0]
  g1[2] = (int)((((unsigned)K >> 16) & 0xffffu)     // tensor_dim0[31:16]
              | (((unsigned)rows & 0xffffu) << 16));// tensor_dim1[15:0]
  g1[3] = (int)(32u << 16);                         // tile_dim0 = 32 elems
  g1[4] = (int)((unsigned)rows & 0xffffu);          // tile_dim1 = rows
  g1[5] = (int)(unsigned)K;                         // tensor_dim0_stride[31:0]
  g1[6] = 0;
  g1[7] = 0;
  i32x4 z4 = {0, 0, 0, 0};                          // 2D: groups 2/3 unused
#if __clang_major__ >= 23
  i32x8 z8 = {0, 0, 0, 0, 0, 0, 0, 0};
  __builtin_amdgcn_tensor_load_to_lds(g0, g1, z4, z4, z8, 0);
#else
  __builtin_amdgcn_tensor_load_to_lds(g0, g1, z4, z4, 0);
#endif
}

__global__ __launch_bounds__(256)
void gemm_bf16_wmma(const u16* __restrict__ A, const u16* __restrict__ Bt,
                    float* __restrict__ outF, u16* __restrict__ outB,
                    const float* __restrict__ bias,
                    int M, int Nn, int K,
                    int addResidual, int doGelu, int remapPatch)
{
  __shared__ __attribute__((aligned(128))) u16 As[2][BM * LSTR]; // 2 x 10KB
  __shared__ __attribute__((aligned(128))) u16 Bs[2][BN * LSTR]; // 2 x 20KB

  const int tid  = threadIdx.x;
  const int wave = tid >> 5;
  const int lane = tid & 31;
  const int bm = blockIdx.y * BM;
  const int bn = blockIdx.x * BN;
  const int wm = (wave & 1) * 64;       // 2 waves along M, 64 rows each
  const int wn = (wave >> 1) * 64;      // 4 waves along N, 64 cols each
  const int hf  = lane >> 4;            // lane half (K split per ISA layout)
  const int l16 = lane & 15;
  const int kb  = hf * 8;

  f32x8 acc[4][4];
#pragma unroll
  for (int i = 0; i < 4; ++i)
#pragma unroll
    for (int j = 0; j < 4; ++j)
#pragma unroll
      for (int e = 0; e < 8; ++e) acc[i][j][e] = 0.0f;

  // prologue: TDM fill of buffer 0 (wave 0 drives the DMA; EXEC-independent)
  if (wave == 0) {
    tdm_load_tile(A,  (unsigned)(uintptr_t)&As[0][0], BM, K, bm, 0);
    tdm_load_tile(Bt, (unsigned)(uintptr_t)&Bs[0][0], BN, K, bn, 0);
  }

  int buf = 0;
  for (int k0 = 0; k0 < K; k0 += BKK) {
    if (wave == 0) __builtin_amdgcn_s_wait_tensorcnt(0);
    __syncthreads();                       // buffer `buf` now filled

    if (wave == 0 && (k0 + BKK) < K) {     // DMA next tile while computing
      tdm_load_tile(A,  (unsigned)(uintptr_t)&As[buf ^ 1][0], BM, K, bm, k0 + BKK);
      tdm_load_tile(Bt, (unsigned)(uintptr_t)&Bs[buf ^ 1][0], BN, K, bn, k0 + BKK);
    }

    const u16* Ab = As[buf];
    const u16* Bb = Bs[buf];
    bf16x16 afr[4], bfr[4];
#pragma unroll
    for (int i = 0; i < 4; ++i)
      afr[i] = load_frag16(Ab + (wm + i * 16 + l16) * LSTR + kb);
#pragma unroll
    for (int j = 0; j < 4; ++j)
      bfr[j] = load_frag16(Bb + (wn + j * 16 + l16) * LSTR + kb);
#pragma unroll
    for (int i = 0; i < 4; ++i)
#pragma unroll
      for (int j = 0; j < 4; ++j)
        acc[i][j] = __builtin_amdgcn_wmma_f32_16x16x32_bf16(
            false, afr[i], false, bfr[j], (short)0, acc[i][j], false, false);
    __syncthreads();                       // all reads of `buf` done
    buf ^= 1;
  }

  // epilogue: lane holds col = l16 within tile, rows hf*8 + r (C layout)
#pragma unroll
  for (int i = 0; i < 4; ++i) {
#pragma unroll
    for (int j = 0; j < 4; ++j) {
      int col = bn + wn + j * 16 + l16;
      float bv = bias ? bias[col] : 0.0f;
#pragma unroll
      for (int r = 0; r < 8; ++r) {
        int row = bm + wm + i * 16 + hf * 8 + r;
        float v = acc[i][j][r] + bv;
        if (doGelu) v = 0.5f * v * (1.0f + erff(v * 0.70710678118654752f));
        size_t orow = remapPatch ? (size_t)((row >> 6) * N_SEQ + (row & 63) + 1)
                                 : (size_t)row;
        size_t idx = orow * (size_t)Nn + col;
        if (outF) {
          if (addResidual) outF[idx] += v; else outF[idx] = v;
        } else {
          outB[idx] = f2bf(v);
        }
      }
    }
  }
}

// ---------------------------- LayerNorm ------------------------------------
__global__ __launch_bounds__(256)
void layernorm_1024(const float* __restrict__ Xin,
                    const float* __restrict__ g, const float* __restrict__ bta,
                    u16* __restrict__ outB, float* __restrict__ outF, int rowMul)
{
  __shared__ float red[256];
  const int r = blockIdx.x;
  const int t = threadIdx.x;
  const float* x = Xin + (size_t)r * rowMul * D_MODEL;
  float4 v = *(const float4*)(x + t * 4);
  red[t] = v.x + v.y + v.z + v.w;
  __syncthreads();
  for (int off = 128; off > 0; off >>= 1) {
    if (t < off) red[t] += red[t + off];
    __syncthreads();
  }
  float mean = red[0] * (1.0f / 1024.0f);
  __syncthreads();
  float dx = v.x - mean, dy = v.y - mean, dz = v.z - mean, dw = v.w - mean;
  red[t] = dx * dx + dy * dy + dz * dz + dw * dw;
  __syncthreads();
  for (int off = 128; off > 0; off >>= 1) {
    if (t < off) red[t] += red[t + off];
    __syncthreads();
  }
  float rstd = rsqrtf(red[0] * (1.0f / 1024.0f) + 1e-5f);
  int c = t * 4;
  float o0 = dx * rstd * g[c + 0] + bta[c + 0];
  float o1 = dy * rstd * g[c + 1] + bta[c + 1];
  float o2 = dz * rstd * g[c + 2] + bta[c + 2];
  float o3 = dw * rstd * g[c + 3] + bta[c + 3];
  size_t base = (size_t)r * D_MODEL + c;
  if (outB) {
    outB[base + 0] = f2bf(o0); outB[base + 1] = f2bf(o1);
    outB[base + 2] = f2bf(o2); outB[base + 3] = f2bf(o3);
  } else {
    outF[base + 0] = o0; outF[base + 1] = o1;
    outF[base + 2] = o2; outF[base + 3] = o3;
  }
}

// ------------------------------ RoPE ---------------------------------------
__global__ __launch_bounds__(256)
void rope_inplace(float* __restrict__ QKV)
{
  size_t idx = (size_t)blockIdx.x * 256 + threadIdx.x;
  int i   = (int)(idx & 15); idx >>= 4;   // freq index 0..15
  int hal = (int)(idx & 1);  idx >>= 1;   // 0: row half, 1: col half
  int h   = (int)(idx & 15); idx >>= 4;   // head
  int t   = (int)(idx & 1);  idx >>= 1;   // 0=q, 1=k
  int row = (int)idx;                     // 0..16639
  int n = row % N_SEQ;
  int pos = 0;
  if (n > 0) { int lin = n - 1; pos = hal ? (lin & 7) : (lin >> 3); }
  float inv = __powf(10000.0f, -(float)i * (1.0f / 16.0f));
  float ang = (float)pos * inv;
  float s, c;
  __sincosf(ang, &s, &c);
  size_t base = (size_t)row * 3072 + (size_t)t * 1024 + h * 64 + hal * 32 + 2 * i;
  float xe = QKV[base], xo = QKV[base + 1];
  QKV[base]     = xe * c - xo * s;
  QKV[base + 1] = xo * c + xe * s;
}

// ---------------------------- Attention ------------------------------------
// one workgroup (4 wave32) per (batch, head); 65x65 scores, wave-shuffle softmax
__global__ __launch_bounds__(128)
void attention_small(const float* __restrict__ QKV, u16* __restrict__ O)
{
  __shared__ float qs[65 * 64];
  __shared__ float ks[65 * 64];
  __shared__ float vs[65 * 64];
  const int b = blockIdx.x >> 4;
  const int h = blockIdx.x & 15;
  const int tid = threadIdx.x;
  const size_t rb = (size_t)b * N_SEQ * 3072;
  for (int e = tid; e < 65 * 64; e += 128) {
    int n = e >> 6, d = e & 63;
    size_t base = rb + (size_t)n * 3072 + h * 64 + d;
    qs[e] = QKV[base];
    ks[e] = QKV[base + 1024];
    vs[e] = QKV[base + 2048];
  }
  __syncthreads();
  const int wv = tid >> 5;
  const int lane = tid & 31;
  for (int i = wv; i < 65; i += 4) {
    const float* qi = qs + i * 64;
    float sv[3];
    float m = -1e30f;
#pragma unroll
    for (int c = 0; c < 3; ++c) {
      int j = c * 32 + lane;
      float s = -1e30f;
      if (j < 65) {
        const float* kj = ks + j * 64;
        float a = 0.0f;
        for (int d = 0; d < 64; ++d) a += qi[d] * kj[d];
        s = a * 0.125f;                       // HD^-0.5
      }
      sv[c] = s;
      m = fmaxf(m, s);
    }
    for (int off = 16; off > 0; off >>= 1) m = fmaxf(m, __shfl_xor(m, off, 32));
    float sum = 0.0f;
#pragma unroll
    for (int c = 0; c < 3; ++c) {
      int j = c * 32 + lane;
      float p = (j < 65) ? __expf(sv[c] - m) : 0.0f;
      sv[c] = p;
      sum += p;
    }
    for (int off = 16; off > 0; off >>= 1) sum += __shfl_xor(sum, off, 32);
    float rs = 1.0f / sum;
    float o0 = 0.0f, o1 = 0.0f;
    for (int j = 0; j < 65; ++j) {
      float pj = __shfl(sv[j >> 5], j & 31, 32) * rs;
      o0 += pj * vs[j * 64 + lane];
      o1 += pj * vs[j * 64 + 32 + lane];
    }
    size_t orow = ((size_t)b * N_SEQ + i) * (size_t)D_MODEL + h * 64;
    O[orow + lane]      = f2bf(o0);
    O[orow + 32 + lane] = f2bf(o1);
  }
}

// --------------------------- small helpers ---------------------------------
__global__ __launch_bounds__(256)
void im2col_patches(const float* __restrict__ img, u16* __restrict__ Ap)
{
  size_t idx = (size_t)blockIdx.x * 256 + threadIdx.x;  // PROWS*1024
  int col = (int)(idx & 1023);
  size_t row = idx >> 10;
  int b = (int)(row >> 6);
  int p = (int)(row & 63);
  int py = col >> 5, px = col & 31;
  int y = (p >> 3) * 32 + py;
  int x = (p & 7) * 32 + px;
  Ap[idx] = f2bf(img[(size_t)b * 65536 + (size_t)y * 256 + x]);
}

__global__ __launch_bounds__(256)
void write_cls(const float* __restrict__ cls, float* __restrict__ X)
{
  int idx = blockIdx.x * 256 + threadIdx.x;   // B_BATCH*1024
  int d = idx & 1023;
  int b = idx >> 10;
  X[(size_t)b * N_SEQ * D_MODEL + d] = cls[d];
}

// W[K x Nn] fp32 -> Wt[Nn x K] bf16 (transposed for WMMA B fragments)
__global__ __launch_bounds__(256)
void wconvert_t(const float* __restrict__ W, u16* __restrict__ Wt, int K, int Nn)
{
  size_t idx = (size_t)blockIdx.x * 256 + threadIdx.x;  // Nn*K
  int k = (int)(idx % (size_t)K);
  size_t n = idx / (size_t)K;
  Wt[idx] = f2bf(W[(size_t)k * Nn + n]);
}

// ------------------------------ launch -------------------------------------
extern "C" void kernel_launch(void* const* d_in, const int* in_sizes, int n_in,
                              void* d_out, int out_size, void* d_ws, size_t ws_size,
                              hipStream_t stream)
{
  (void)in_sizes; (void)n_in; (void)out_size; (void)ws_size;
  const float* img     = (const float*)d_in[0];
  const float* patch_W = (const float*)d_in[1];
  const float* patch_b = (const float*)d_in[2];
  const float* cls_tok = (const float*)d_in[3];
  const float* ln1_g   = (const float*)d_in[4];
  const float* ln1_b   = (const float*)d_in[5];
  const float* Wqkv    = (const float*)d_in[6];
  const float* Wproj   = (const float*)d_in[7];
  const float* bproj   = (const float*)d_in[8];
  const float* ln2_g   = (const float*)d_in[9];
  const float* ln2_b   = (const float*)d_in[10];
  const float* W1      = (const float*)d_in[11];
  const float* b1      = (const float*)d_in[12];
  const float* W2      = (const float*)d_in[13];
  const float* b2      = (const float*)d_in[14];
  const float* lnf_g   = (const float*)d_in[15];
  const float* lnf_b   = (const float*)d_in[16];

  uint8_t* wp = (uint8_t*)d_ws;
  auto carve = [&](size_t bytes) -> void* {
    void* r = (void*)wp;
    wp += (bytes + 255) & ~(size_t)255;
    return r;
  };
  float* X     = (float*)carve((size_t)ROWS * D_MODEL * 4);       // residual stream
  u16*   Hb    = (u16*)  carve((size_t)ROWS * D_MODEL * 2);       // LN out (bf16)
  float* QKV   = (float*)carve((size_t)ROWS * 3 * D_MODEL * 4);
  u16*   ATTO  = (u16*)  carve((size_t)ROWS * D_MODEL * 2);
  u16*   MID   = (u16*)  carve((size_t)ROWS * FF * 2);
  u16*   APAT  = (u16*)  carve((size_t)PROWS * D_MODEL * 2);
  u16*   PWt   = (u16*)  carve((size_t)D_MODEL * D_MODEL * 2);
  u16*   WqkvT = (u16*)  carve((size_t)3 * D_MODEL * D_MODEL * 2);
  u16*   WprojT= (u16*)  carve((size_t)D_MODEL * D_MODEL * 2);
  u16*   W1T   = (u16*)  carve((size_t)FF * D_MODEL * 2);
  u16*   W2T   = (u16*)  carve((size_t)D_MODEL * FF * 2);

  dim3 blk(256);

  im2col_patches<<<(PROWS * D_MODEL) / 256, blk, 0, stream>>>(img, APAT);
  wconvert_t<<<(D_MODEL * D_MODEL) / 256, blk, 0, stream>>>(patch_W, PWt, D_MODEL, D_MODEL);
  write_cls<<<(B_BATCH * D_MODEL) / 256, blk, 0, stream>>>(cls_tok, X);
  // patch embed: X[b*65 + p + 1, :] = patches @ patch_W + patch_b
  gemm_bf16_wmma<<<dim3(D_MODEL / BN, PROWS / BM), blk, 0, stream>>>(
      APAT, PWt, X, nullptr, patch_b, PROWS, D_MODEL, D_MODEL, 0, 0, 1);

  for (int l = 0; l < DEPTH; ++l) {
    wconvert_t<<<(3 * D_MODEL * D_MODEL) / 256, blk, 0, stream>>>(
        Wqkv + (size_t)l * D_MODEL * 3 * D_MODEL, WqkvT, D_MODEL, 3 * D_MODEL);
    wconvert_t<<<(D_MODEL * D_MODEL) / 256, blk, 0, stream>>>(
        Wproj + (size_t)l * D_MODEL * D_MODEL, WprojT, D_MODEL, D_MODEL);
    wconvert_t<<<(FF * D_MODEL) / 256, blk, 0, stream>>>(
        W1 + (size_t)l * D_MODEL * FF, W1T, D_MODEL, FF);
    wconvert_t<<<(D_MODEL * FF) / 256, blk, 0, stream>>>(
        W2 + (size_t)l * FF * D_MODEL, W2T, FF, D_MODEL);

    layernorm_1024<<<ROWS, blk, 0, stream>>>(
        X, ln1_g + l * D_MODEL, ln1_b + l * D_MODEL, Hb, nullptr, 1);
    gemm_bf16_wmma<<<dim3(3 * D_MODEL / BN, ROWS / BM), blk, 0, stream>>>(
        Hb, WqkvT, QKV, nullptr, nullptr, ROWS, 3 * D_MODEL, D_MODEL, 0, 0, 0);
    rope_inplace<<<((size_t)ROWS * HEADS * 2 * 2 * 16) / 256, blk, 0, stream>>>(QKV);
    attention_small<<<B_BATCH * HEADS, dim3(128), 0, stream>>>(QKV, ATTO);
    gemm_bf16_wmma<<<dim3(D_MODEL / BN, ROWS / BM), blk, 0, stream>>>(
        ATTO, WprojT, X, nullptr, bproj + l * D_MODEL, ROWS, D_MODEL, D_MODEL, 1, 0, 0);
    layernorm_1024<<<ROWS, blk, 0, stream>>>(
        X, ln2_g + l * D_MODEL, ln2_b + l * D_MODEL, Hb, nullptr, 1);
    gemm_bf16_wmma<<<dim3(FF / BN, ROWS / BM), blk, 0, stream>>>(
        Hb, W1T, nullptr, MID, b1 + l * FF, ROWS, FF, D_MODEL, 0, 1, 0);
    gemm_bf16_wmma<<<dim3(D_MODEL / BN, ROWS / BM), blk, 0, stream>>>(
        MID, W2T, X, nullptr, b2 + l * D_MODEL, ROWS, D_MODEL, FF, 1, 0, 0);
  }

  // final LN over CLS rows -> d_out (256 x 1024 fp32)
  layernorm_1024<<<B_BATCH, blk, 0, stream>>>(
      X, lnf_g, lnf_b, nullptr, (float*)d_out, N_SEQ);
}
